// DQN_37142877176176
// MI455X (gfx1250) — matrix operand order repaired and checked by reading, not compiled, not run on hardware
//
#include <hip/hip_runtime.h>
#include <hip/hip_bf16.h>
#include <cstdint>
#include <cstddef>

typedef _Float16 half_t;
typedef __attribute__((ext_vector_type(16))) _Float16 v16h;
typedef __attribute__((ext_vector_type(8)))  _Float16 v8h;
typedef __attribute__((ext_vector_type(2)))  _Float16 v2h;
typedef __attribute__((ext_vector_type(8)))  float    v8f;

#define TSTEPS 1024
#define HID    512
#define NACT   23
#define GW_    79
#define GH_    21
#define GP_    (GW_*GH_)      // 1659
#define CW_    9
#define CH_    9
#define CP_    (CW_*CH_)      // 81
#define FLATSZ 13920
#define GFLAT  13272          // 8*1659
#define NLSTMBLK 16
#define GWAVES 4              // waves per GEMM block

// ---------------------------------------------------------------------------
// prep: int glyphs -> channel-minor f16 activations [T][P][16] (ch0=val, rest 0)
// ---------------------------------------------------------------------------
__global__ void prep_act(const int* __restrict__ src, half_t* __restrict__ dst, int n) {
  int i = blockIdx.x * 256 + threadIdx.x;
  if (i >= n) return;
  v16h o = {};
  o[0] = (half_t)((float)src[i] * (1.0f / 5991.0f));
  *(v16h*)(dst + (size_t)i * 16) = o;
}

// ---------------------------------------------------------------------------
// prep: rearrange conv weights [cout][cin][3][3] f32 -> [16co][10tap][16ci] f16
// ---------------------------------------------------------------------------
struct ConvWArgs { const float* w[10]; const float* b[10]; };

__global__ void prep_convw(ConvWArgs args, half_t* __restrict__ wq, float* __restrict__ bq) {
  const int cins[5]  = {1, 16, 16, 16, 16};
  const int couts[5] = {16, 16, 16, 16, 8};
  for (int i = blockIdx.x * 256 + threadIdx.x; i < 10 * 2560; i += gridDim.x * 256) {
    int l   = i / 2560;
    int rr  = i % 2560;
    int co  = rr / 160;
    int tap = (rr / 16) % 10;
    int ci  = rr % 16;
    int cin = cins[l % 5], cout = couts[l % 5];
    float v = 0.f;
    if (co < cout && ci < cin && tap < 9)
      v = args.w[l][(co * cin + ci) * 9 + tap];
    wq[i] = (half_t)v;
  }
  for (int i = blockIdx.x * 256 + threadIdx.x; i < 160; i += gridDim.x * 256) {
    int l = i / 16, co = i % 16;
    bq[i] = (co < couts[l % 5]) ? args.b[l][co] : 0.f;
  }
}

__global__ void f32_to_f16(const float* __restrict__ s, half_t* __restrict__ d, int n) {
  int i = blockIdx.x * 256 + threadIdx.x;
  if (i < n) d[i] = (half_t)s[i];
}

__global__ void add_vec(const float* __restrict__ a, const float* __restrict__ b,
                        float* __restrict__ o, int n) {
  int i = blockIdx.x * 256 + threadIdx.x;
  if (i < n) o[i] = a[i] + b[i];
}

// ---------------------------------------------------------------------------
// 3x3 SAME conv + bias + relu as WMMA: 5 x v_wmma_f32_16x16x32_f16 per
// 16-pixel tile. K=32 chunk j covers taps (2j, 2j+1) x 16 input channels.
// ---------------------------------------------------------------------------
template <int W_, int H_>
__global__ void __launch_bounds__(256) conv_wmma(
    const half_t* __restrict__ in,   // [T][P][16]
    half_t* __restrict__ out,        // [T][P][16]
    const half_t* __restrict__ wq,   // [16co][10tap][16ci]
    const float* __restrict__ bias)  // [16]
{
  constexpr int P  = W_ * H_;
  constexpr int NT = (P + 15) / 16;
  const int t    = blockIdx.x;
  const int lane = threadIdx.x & 31;
  const int wv   = threadIdx.x >> 5;
  const int m    = lane & 15;
  const int hs8  = lane >> 4;

  v16h afrag[5];
#pragma unroll
  for (int j = 0; j < 5; ++j) {
#pragma unroll
    for (int e = 0; e < 16; ++e) {
      int tap = 2 * j + hs8;
      int ci  = e;
      afrag[j][e] = wq[(m * 10 + tap) * 16 + ci];
    }
  }
  float bv[8];
#pragma unroll
  for (int r = 0; r < 8; ++r) bv[r] = bias[r + hs8 * 8];

  const half_t* inT  = in  + (size_t)t * P * 16;
  half_t*       outT = out + (size_t)t * P * 16;

  for (int tile = wv; tile < NT; tile += 8) {
    const int p = tile * 16 + m;
    const int y = p / W_, x = p % W_;
    v8f acc = {};
#pragma unroll
    for (int j = 0; j < 5; ++j) {
      const int tap = 2 * j + hs8;
      v16h b = {};
      if (tap < 9 && p < P) {
        const int iy = y + tap / 3 - 1;
        const int ix = x + tap % 3 - 1;
        if (iy >= 0 && iy < H_ && ix >= 0 && ix < W_)
          b = *(const v16h*)(inT + (size_t)(iy * W_ + ix) * 16);
      }
      acc = __builtin_amdgcn_wmma_f32_16x16x32_f16(false, afrag[j], false, b,
                                                   (short)0, acc, false, false);
    }
    if (p < P) {
      v8h o;
#pragma unroll
      for (int r = 0; r < 8; ++r) {
        float v = acc[r] + bv[r];
        v = v > 0.f ? v : 0.f;
        o[r] = (half_t)v;
      }
      *(v8h*)(outT + (size_t)p * 16 + hs8 * 8) = o;
    }
  }
}

// ---------------------------------------------------------------------------
// flatten: x[t][k] = concat(glyph acts ch-major, crop acts ch-major)
// ---------------------------------------------------------------------------
__global__ void pack_x(const half_t* __restrict__ ag, const half_t* __restrict__ ac,
                       half_t* __restrict__ x, int n) {
  int i = blockIdx.x * 256 + threadIdx.x;
  if (i >= n) return;
  int t = i / FLATSZ, k = i % FLATSZ;
  half_t v;
  if (k < GFLAT) {
    int ch = k / GP_, p = k % GP_;
    v = ag[((size_t)t * GP_ + p) * 16 + ch];
  } else {
    int kk = k - GFLAT;
    int ch = kk / CP_, p = kk % CP_;
    v = ac[((size_t)t * CP_ + p) * 16 + ch];
  }
  x[(size_t)t * FLATSZ + k] = v;
}

// ---------------------------------------------------------------------------
// WMMA GEMM: out[M][N] = act?(A[M][K] @ Wt[N][K]^T + bias[N])
// One 16x64 output tile per wave (A fragment reused 4x). The 4 B sub-tiles
// (64 cols x 32 k = 4KB) are staged into LDS per K-chunk with double-buffered
// global_load_async_to_lds_b128 (ASYNCcnt), drained via s_wait_asynccnt.
// ---------------------------------------------------------------------------
template <bool RELU, bool OUTF16>
__global__ void __launch_bounds__(128) gemm_wmma(
    const half_t* __restrict__ A, const half_t* __restrict__ Wt,
    const float* __restrict__ bias, void* __restrict__ outp,
    int M, int N, int K)
{
  __shared__ __attribute__((aligned(128))) half_t bstage[GWAVES][2][64 * 32];
  const int wv      = threadIdx.x >> 5;
  const int wid     = blockIdx.x * GWAVES + wv;
  const int ntile64 = N >> 6;
  const int tiles   = (M >> 4) * ntile64;
  if (wid >= tiles) return;
  const int mt   = wid / ntile64;
  const int nt0  = (wid % ntile64) << 2;      // first of 4 consecutive 16-col tiles
  const int lane = threadIdx.x & 31;
  const int l16  = lane & 15;
  const int hs8  = lane >> 4;
  const int row  = mt * 16 + l16;
  const int col0 = nt0 * 16;

  const half_t* pa = A + (size_t)row * K;
  const uint64_t wtbase = (uint64_t)(uintptr_t)Wt;
  const uint32_t lb0 = (uint32_t)(uintptr_t)&bstage[wv][0][0];
  const uint32_t lb1 = (uint32_t)(uintptr_t)&bstage[wv][1][0];

  // stage 64 cols x 64B of Wt for K-chunk kc: 256 16B chunks -> 8 async b128 ops
  auto issue = [&](int kc, int buf) {
    const uint32_t lb = buf ? lb1 : lb0;
#pragma unroll
    for (int q = 0; q < 8; ++q) {
      int chunk = q * 32 + lane;
      int c = chunk >> 2, part = chunk & 3;
      uint64_t ga = wtbase +
                    ((uint64_t)(col0 + c) * (uint64_t)K + (uint64_t)kc * 32) * 2 +
                    (uint64_t)part * 16;
      uint32_t lo = lb + (uint32_t)chunk * 16;
      asm volatile("global_load_async_to_lds_b128 %0, %1, off"
                   :: "v"(lo), "v"(ga) : "memory");
    }
  };

  v8f acc[4] = {{}, {}, {}, {}};
  const int KC = K >> 5;
  issue(0, 0);
  for (int kc = 0; kc < KC; ++kc) {
    const int buf = kc & 1;
    if (kc + 1 < KC) {
      issue(kc + 1, buf ^ 1);
      asm volatile("s_wait_asynccnt 0x8" ::: "memory");   // previous batch done
    } else {
      asm volatile("s_wait_asynccnt 0x0" ::: "memory");
    }
    union { v16h v; v8h h[2]; } a;
    a.h[0] = *(const v8h*)(pa + kc * 32 + hs8 * 16);
    a.h[1] = *(const v8h*)(pa + kc * 32 + hs8 * 16 + 8);
    __builtin_prefetch(pa + kc * 32 + 256, 0, 0);         // global_prefetch_b8
    const half_t* lb = &bstage[wv][buf][0];
#pragma unroll
    for (int tt = 0; tt < 4; ++tt) {
      v16h b = *(const v16h*)(lb + (tt * 16 + l16) * 32 + hs8 * 16);
      acc[tt] = __builtin_amdgcn_wmma_f32_16x16x32_f16(false, a.v, false, b,
                                                       (short)0, acc[tt], false, false);
    }
  }
#pragma unroll
  for (int tt = 0; tt < 4; ++tt) {
    const int col  = col0 + tt * 16 + l16;
    const float bc = bias[col];
#pragma unroll
    for (int r = 0; r < 8; ++r) {
      int mr  = mt * 16 + r + hs8 * 8;
      float v = acc[tt][r] + bc;
      if (RELU) v = fmaxf(v, 0.f);
      if (OUTF16) ((half_t*)outp)[(size_t)mr * N + col] = (half_t)v;
      else        ((float*)outp)[(size_t)mr * N + col] = v;
    }
  }
}

// ---------------------------------------------------------------------------
// LSTM recurrence: 16 persistent blocks, 128 whh rows each (f16, L2-resident),
// packed-f16 dot2 matvec, one device barrier/step, double-buffered gates.
// ---------------------------------------------------------------------------
__device__ __forceinline__ float sigf(float x) { return 1.f / (1.f + __expf(-x)); }

__global__ void __launch_bounds__(256) lstm_kernel(
    const float* __restrict__ gatesx,  // [T][2048]  (x@wih.T + bih + bhh)
    const half_t* __restrict__ whh,    // [2048][512] f16
    const int* __restrict__ done,
    float* __restrict__ hs,            // [T][512]
    float* __restrict__ gbuf,          // [2][2048]
    int* __restrict__ bar)
{
  __shared__ float    h_s[HID];
  __shared__ float    c_s[HID];
  __shared__ float    red[256];
  __shared__ uint32_t hpk[HID / 2];    // h as packed f16 pairs
  const int tid  = threadIdx.x;
  const int blk  = blockIdx.x;
  const int base = blk * 128;
  h_s[2 * tid] = 0.f; h_s[2 * tid + 1] = 0.f;
  c_s[2 * tid] = 0.f; c_s[2 * tid + 1] = 0.f;
  hpk[tid] = 0u;
  __syncthreads();

  const int r     = tid >> 1;
  const int khalf = tid & 1;
  const half_t* wrow = whh + (size_t)(base + r) * HID + khalf * 256;

  int gen = 0;
  for (int t = 0; t < TSTEPS; ++t) {
    const float nd = 1.f - (float)done[t];
    {  // reset-scale h,c; repack h to f16 pairs
      float h0 = h_s[2 * tid] * nd, h1 = h_s[2 * tid + 1] * nd;
      h_s[2 * tid] = h0; h_s[2 * tid + 1] = h1;
      c_s[2 * tid] *= nd; c_s[2 * tid + 1] *= nd;
      union { uint32_t u; v2h v; } pk;
      pk.v[0] = (half_t)h0; pk.v[1] = (half_t)h1;
      hpk[tid] = pk.u;
    }
    __syncthreads();

    // 128x512 matvec slice: 2 threads/row, 256 MACs each via v_dot2_f32_f16
    float acc = 0.f;
#if __has_builtin(__builtin_amdgcn_fdot2)
    {
      const v2h* hp2 = (const v2h*)hpk;
#pragma unroll 4
      for (int kb = 0; kb < 32; ++kb) {
        uint4 u = ((const uint4*)wrow)[kb];
        const v2h* wp = (const v2h*)&u;
        const int p0 = khalf * 128 + kb * 4;
#pragma unroll
        for (int q = 0; q < 4; ++q)
          acc = __builtin_amdgcn_fdot2(wp[q], hp2[p0 + q], acc, false);
      }
    }
#else
#pragma unroll 4
    for (int kb = 0; kb < 32; ++kb) {
      uint4 u = ((const uint4*)wrow)[kb];
      const half_t* hp = (const half_t*)&u;
      const int k0 = khalf * 256 + kb * 8;
#pragma unroll
      for (int q = 0; q < 8; ++q) acc += (float)hp[q] * h_s[k0 + q];
    }
#endif
    red[tid] = acc;
    __syncthreads();

    float* gout = gbuf + (size_t)(t & 1) * 2048;
    if (khalf == 0)
      gout[base + r] = red[tid] + red[tid + 1] + gatesx[(size_t)t * 2048 + base + r];

    // device-scope barrier (monotonic counter + s_sleep spin)
    ++gen;
    __syncthreads();
    if (tid == 0) {
      __threadfence();
      atomicAdd(bar, 1);
      while (__hip_atomic_load(bar, __ATOMIC_RELAXED, __HIP_MEMORY_SCOPE_AGENT) <
             gen * NLSTMBLK)
        __builtin_amdgcn_s_sleep(1);
      __threadfence();
    }
    __syncthreads();

    // redundant elementwise update (every block needs next h)
#pragma unroll
    for (int s = 0; s < 2; ++s) {
      const int j = 2 * tid + s;
      float gi = gout[j];
      float gf = gout[HID + j];
      float gg = gout[2 * HID + j];
      float go = gout[3 * HID + j];
      float cs = sigf(gf) * c_s[j] + sigf(gi) * tanhf(gg);
      float hn = sigf(go) * tanhf(cs);
      c_s[j] = cs;
      h_s[j] = hn;
      if (blk == 0) hs[(size_t)t * HID + j] = hn;
    }
    __syncthreads();
  }
}

// ---------------------------------------------------------------------------
// policy head: out[T][23] = hs @ l3w.T + l3b
// ---------------------------------------------------------------------------
__global__ void l3_kernel(const float* __restrict__ hsv, const float* __restrict__ w,
                          const float* __restrict__ b, float* __restrict__ out) {
  const int t = blockIdx.x;
  const int a = threadIdx.x;
  if (a >= NACT) return;
  const float* hrow = hsv + (size_t)t * HID;
  const float* wrow = w + (size_t)a * HID;
  float acc = b[a];
  for (int k = 0; k < HID; ++k) acc += hrow[k] * wrow[k];
  out[(size_t)t * NACT + a] = acc;
}

// ---------------------------------------------------------------------------
// host side
// ---------------------------------------------------------------------------
static inline size_t alignup(size_t x) { return (x + 255) & ~(size_t)255; }

extern "C" void kernel_launch(void* const* d_in, const int* in_sizes, int n_in,
                              void* d_out, int out_size, void* d_ws, size_t ws_size,
                              hipStream_t stream) {
  (void)in_sizes; (void)n_in; (void)out_size; (void)ws_size;
  const int*   glyphs = (const int*)d_in[0];
  const int*   crop   = (const int*)d_in[1];
  const int*   done   = (const int*)d_in[2];
  const float* l1w = (const float*)d_in[23];
  const float* l1b = (const float*)d_in[24];
  const float* l2w = (const float*)d_in[25];
  const float* l2b = (const float*)d_in[26];
  const float* l3w = (const float*)d_in[27];
  const float* l3b = (const float*)d_in[28];
  const float* wih = (const float*)d_in[29];
  const float* whh = (const float*)d_in[30];
  const float* bih = (const float*)d_in[31];
  const float* bhh = (const float*)d_in[32];

  char* ws = (char*)d_ws;
  size_t off = 0;
  auto alloc = [&](size_t bytes) { void* p = ws + off; off += alignup(bytes); return p; };

  half_t* actg0  = (half_t*)alloc((size_t)TSTEPS * GP_ * 16 * 2);
  half_t* actg1  = (half_t*)alloc((size_t)TSTEPS * GP_ * 16 * 2);
  half_t* actc0  = (half_t*)alloc((size_t)TSTEPS * CP_ * 16 * 2);
  half_t* actc1  = (half_t*)alloc((size_t)TSTEPS * CP_ * 16 * 2);
  half_t* xpack  = (half_t*)alloc((size_t)TSTEPS * FLATSZ * 2);
  half_t* l1w16  = (half_t*)alloc((size_t)HID * FLATSZ * 2);
  half_t* l2w16  = (half_t*)alloc((size_t)HID * HID * 2);
  half_t* wih16  = (half_t*)alloc((size_t)4 * HID * HID * 2);
  half_t* whh16  = (half_t*)alloc((size_t)4 * HID * HID * 2);
  half_t* convw  = (half_t*)alloc(10 * 2560 * 2);
  float*  convb  = (float*)alloc(160 * 4);
  float*  bihhh  = (float*)alloc(4 * HID * 4);
  half_t* y16    = (half_t*)alloc((size_t)TSTEPS * HID * 2);
  half_t* z16    = (half_t*)alloc((size_t)TSTEPS * HID * 2);
  float*  gatesx = (float*)alloc((size_t)TSTEPS * 4 * HID * 4);
  float*  gbuf   = (float*)alloc(2 * 4 * HID * 4);
  float*  hsbuf  = (float*)alloc((size_t)TSTEPS * HID * 4);
  int*    bar    = (int*)alloc(256);

  hipMemsetAsync(bar, 0, 256, stream);

  {
    int n = TSTEPS * GP_;
    prep_act<<<(n + 255) / 256, 256, 0, stream>>>(glyphs, actg0, n);
    n = TSTEPS * CP_;
    prep_act<<<(n + 255) / 256, 256, 0, stream>>>(crop, actc0, n);
  }
  {
    ConvWArgs a;
    for (int i = 0; i < 5; ++i) { a.w[i]     = (const float*)d_in[3 + 2 * i];
                                  a.b[i]     = (const float*)d_in[4 + 2 * i];
                                  a.w[5 + i] = (const float*)d_in[13 + 2 * i];
                                  a.b[5 + i] = (const float*)d_in[14 + 2 * i]; }
    prep_convw<<<16, 256, 0, stream>>>(a, convw, convb);
    int n = HID * FLATSZ;
    f32_to_f16<<<(n + 255) / 256, 256, 0, stream>>>(l1w, l1w16, n);
    n = HID * HID;
    f32_to_f16<<<(n + 255) / 256, 256, 0, stream>>>(l2w, l2w16, n);
    n = 4 * HID * HID;
    f32_to_f16<<<(n + 255) / 256, 256, 0, stream>>>(wih, wih16, n);
    f32_to_f16<<<(n + 255) / 256, 256, 0, stream>>>(whh, whh16, n);
    add_vec<<<(4 * HID + 255) / 256, 256, 0, stream>>>(bih, bhh, bihhh, 4 * HID);
  }
  {
    half_t* gio[6] = {actg0, actg1, actg0, actg1, actg0, actg1};
    half_t* cio[6] = {actc0, actc1, actc0, actc1, actc0, actc1};
    for (int l = 0; l < 5; ++l) {
      conv_wmma<GW_, GH_><<<TSTEPS, 256, 0, stream>>>(gio[l], gio[l + 1],
                                                      convw + l * 2560, convb + l * 16);
      conv_wmma<CW_, CH_><<<TSTEPS, 256, 0, stream>>>(cio[l], cio[l + 1],
                                                      convw + (5 + l) * 2560,
                                                      convb + (5 + l) * 16);
    }
  }
  {
    int n = TSTEPS * FLATSZ;
    pack_x<<<(n + 255) / 256, 256, 0, stream>>>(actg1, actc1, xpack, n);

    int tiles = (TSTEPS / 16) * (HID / 64);            // 512
    gemm_wmma<true, true><<<(tiles + GWAVES - 1) / GWAVES, 128, 0, stream>>>(
        xpack, l1w16, l1b, y16, TSTEPS, HID, FLATSZ);
    gemm_wmma<true, true><<<(tiles + GWAVES - 1) / GWAVES, 128, 0, stream>>>(
        y16, l2w16, l2b, z16, TSTEPS, HID, HID);
    tiles = (TSTEPS / 16) * (4 * HID / 64);            // 2048
    gemm_wmma<false, false><<<(tiles + GWAVES - 1) / GWAVES, 128, 0, stream>>>(
        z16, wih16, bihhh, gatesx, TSTEPS, 4 * HID, HID);
  }
  lstm_kernel<<<NLSTMBLK, 256, 0, stream>>>(gatesx, whh16, done, hsbuf, gbuf, bar);
  l3_kernel<<<TSTEPS, 32, 0, stream>>>(hsbuf, l3w, l3b, (float*)d_out);
}
